// CrissCrossAttention_fake_x_84189948936441
// MI455X (gfx1250) — matrix operand order repaired and checked by reading, not compiled, not run on hardware
//
#include <hip/hip_runtime.h>
#include <hip/hip_bf16.h>

// CDNA5 / gfx1250: wave32, WMMA 16x16x32 bf16 (D,C = f32)
typedef __attribute__((ext_vector_type(16))) __bf16 v16bf;
typedef __attribute__((ext_vector_type(8)))  float  v8f;

#define NB  4
#define NC  256
#define NCQ 32
#define NH  64
#define NW  64
#define NHW 4096

__device__ __forceinline__ v8f wmma_bf16(v16bf a, v16bf b, v8f c) {
    // args: (neg_a, A, neg_b, B, c_mod, C, reuse_a, reuse_b)
    return __builtin_amdgcn_wmma_f32_16x16x32_bf16(false, a, false, b, (short)0, c, false, false);
}

// ds_swizzle_b32 XOR pattern within 32 lanes: offset15=0, and=0x1f, or=0, xor=k
#define SWZ_XOR(v, k) \
    __int_as_float(__builtin_amdgcn_ds_swizzle(__float_as_int(v), 0x7C00 | (k)))

// ---- WMMA fragment element positions within a 512-element (32 lane x 16) tile ----
// A (16x32, 16-bit): lane = M + 16*(K>>4); j = K&15
__device__ __forceinline__ int afrag_pos(int mrow, int k) {
    int lane = mrow + ((k >> 4) << 4);
    int j    = k & 15;
    return lane * 16 + j;
}
// B (32x16, 16-bit): lanes 0-15: K=0..15 (j=K), N=lane; lanes 16-31: K=16..31, N=lane-16
__device__ __forceinline__ int bfrag_pos(int ncol, int k) {
    int lane = ncol + ((k >> 4) << 4);
    int j    = k & 15;
    return lane * 16 + j;
}

// ================= prep kernels: build bf16 fragments in workspace ===============

// Q: A-fragments. qA[b][qt(256)][512], entry (M=m&15, K=cq) = wq·x + bq
__global__ __launch_bounds__(256) void k_prep_q(const float* __restrict__ x,
        const float* __restrict__ wq, const float* __restrict__ bq,
        __bf16* __restrict__ qA) {
    int tid = blockIdx.x * 256 + threadIdx.x;        // NB*NCQ*NHW = 524288
    int b  = tid >> 17;
    int r  = tid & 131071;
    int cq = r >> 12;
    int m  = r & 4095;
    const float* xr = x + ((size_t)b * NC) * NHW + m;
    const float* wr = wq + cq * NC;
    float acc = bq[cq];
#pragma unroll 4
    for (int ci = 0; ci < NC; ++ci) acc += wr[ci] * xr[(size_t)ci * NHW];
    size_t base = (size_t)(b * 256 + (m >> 4)) * 512;
    qA[base + afrag_pos(m & 15, cq)] = (__bf16)acc;
}

// K: B-fragments. kB[b][kt(256)][512], entry (N=mkey&15, K=cq)
__global__ __launch_bounds__(256) void k_prep_k(const float* __restrict__ x,
        const float* __restrict__ wk, const float* __restrict__ bk,
        __bf16* __restrict__ kB) {
    int tid = blockIdx.x * 256 + threadIdx.x;        // 524288
    int b  = tid >> 17;
    int r  = tid & 131071;
    int cq = r >> 12;
    int m  = r & 4095;
    const float* xr = x + ((size_t)b * NC) * NHW + m;
    const float* wr = wk + cq * NC;
    float acc = bk[cq];
#pragma unroll 4
    for (int ci = 0; ci < NC; ++ci) acc += wr[ci] * xr[(size_t)ci * NHW];
    size_t base = (size_t)(b * 256 + (m >> 4)) * 512;
    kB[base + bfrag_pos(m & 15, cq)] = (__bf16)acc;
}

// V: A-fragments, two copies: vA (plain key index) and vTA (spatially transposed key)
// vA[b][ct(16)][nk(128)][512], entry (M=c&15, K=n&31) = v[c, n]
__global__ __launch_bounds__(256) void k_prep_v(const float* __restrict__ x,
        const float* __restrict__ wv, const float* __restrict__ bv,
        __bf16* __restrict__ vA, __bf16* __restrict__ vTA) {
    int tid = blockIdx.x * 256 + threadIdx.x;        // NB*NC*NHW = 4194304
    int b = tid >> 20;
    int r = tid & 1048575;
    int c = r >> 12;
    int n = r & 4095;
    const float* xr = x + ((size_t)b * NC) * NHW + n;
    const float* wr = wv + c * NC;
    float acc = bv[c];
#pragma unroll 4
    for (int ci = 0; ci < NC; ++ci) acc += wr[ci] * xr[(size_t)ci * NHW];
    __bf16 v = (__bf16)acc;
    size_t tA = ((size_t)((b * 16 + (c >> 4)) * 128 + (n >> 5))) * 512
              + afrag_pos(c & 15, n & 31);
    vA[tA] = v;
    int nt = (n & 63) * 64 + (n >> 6);               // transpose is an involution
    size_t tT = ((size_t)((b * 16 + (c >> 4)) * 128 + (nt >> 5))) * 512
              + afrag_pos(c & 15, nt & 31);
    vTA[tT] = v;
}

// wo: A-fragments. woA[ot(16)][kk(32)][512], entry (M=o&15, K=cc&31) = wo[o, cc]
__global__ __launch_bounds__(256) void k_prep_wo(const float* __restrict__ wo,
        __bf16* __restrict__ woA) {
    int tid = blockIdx.x * 256 + threadIdx.x;        // 256*1024 = 262144
    int o  = tid >> 10;
    int cc = tid & 1023;
    size_t t = ((size_t)((o >> 4) * 32 + (cc >> 5))) * 512 + afrag_pos(o & 15, cc & 31);
    woA[t] = (__bf16)wo[o * 1024 + cc];
}

// ============== energy + softmax: one wave per (b, query-tile, h) ================
// energy[n, key] = sum_c q[c,n] k[c,key]; softmax over w (64 contiguous keys).
// Writes two layouts:
//   att [b][n][key]              (B-fragment-contiguous for p_h / p_v)
//   attP[b][n_w][h][w][n_h]      (B-fragment-contiguous for p_d / p_a; n = n_h*64+n_w,
//                                 key = h*64+w, n_h innermost)
__global__ __launch_bounds__(256) void k_energy_softmax(const __bf16* __restrict__ qA,
        const __bf16* __restrict__ kB, __bf16* __restrict__ att,
        __bf16* __restrict__ attP) {
    int wid  = threadIdx.x >> 5;
    int lane = threadIdx.x & 31;
    int gw = blockIdx.x * 8 + wid;                   // NB*256*64 = 65536 waves
    int b  = gw >> 14;
    int r  = gw & 16383;
    int qt = r >> 6;
    int h  = r & 63;

    v16bf qf = *(const v16bf*)(qA + ((size_t)(b * 256 + qt) * 32 + lane) * 16);
    v8f e[4];
#pragma unroll
    for (int t = 0; t < 4; ++t) {
        int kt = h * 4 + t;
        v16bf kf = *(const v16bf*)(kB + ((size_t)(b * 256 + kt) * 32 + lane) * 16);
        v8f z = {};
        e[t] = wmma_bf16(qf, kf, z);                 // one wmma covers full CQ=32
    }
    int hi   = lane >> 4;
    int lcol = lane & 15;
    const size_t battP = (size_t)b * NHW * NHW;
#pragma unroll
    for (int rr = 0; rr < 8; ++rr) {
        // row-wise max over 64 keys: 4 accs per lane + XOR-swizzle over 16-lane half
        float mx = fmaxf(fmaxf(e[0][rr], e[1][rr]), fmaxf(e[2][rr], e[3][rr]));
        mx = fmaxf(mx, SWZ_XOR(mx, 1));
        mx = fmaxf(mx, SWZ_XOR(mx, 2));
        mx = fmaxf(mx, SWZ_XOR(mx, 4));
        mx = fmaxf(mx, SWZ_XOR(mx, 8));
        float s = 0.f;
#pragma unroll
        for (int t = 0; t < 4; ++t) { float ev = __expf(e[t][rr] - mx); e[t][rr] = ev; s += ev; }
        s += SWZ_XOR(s, 1);
        s += SWZ_XOR(s, 2);
        s += SWZ_XOR(s, 4);
        s += SWZ_XOR(s, 8);
        float inv = 1.f / s;
        int n  = qt * 16 + rr + 8 * hi;              // C/D layout: vgpr rr -> row rr(+8)
        int nh = n >> 6, nw = n & 63;
        size_t rowb = ((size_t)b * NHW + n) * NHW + h * 64 + lcol;
        size_t prow = battP + ((size_t)(nw * 64 + h) * 64) * 64 + nh;  // + w*64 added per t
#pragma unroll
        for (int t = 0; t < 4; ++t) {
            __bf16 a = (__bf16)(e[t][rr] * inv);
            int w = t * 16 + lcol;
            att[rowb + t * 16]       = a;
            attP[prow + (size_t)w * 64] = a;
        }
    }
}

// ====== aggregation: 4 variants fused; one wave per (b, c-tile, m-tile) ==========
// P[c,m] = sum_key Vx[c,key] * att_perm[key,m]; 128 K-chunks x 4 wmma each.
// All B-fragments are now single contiguous 32B v16bf loads.
__global__ __launch_bounds__(256) void k_aggregate(const __bf16* __restrict__ vA,
        const __bf16* __restrict__ vTA, const __bf16* __restrict__ att,
        const __bf16* __restrict__ attP, __bf16* __restrict__ PT) {
    int wid  = threadIdx.x >> 5;
    int lane = threadIdx.x & 31;
    int gw = blockIdx.x * 8 + wid;                   // NB*16*256 = 16384 waves
    int b  = gw >> 12;
    int r  = gw & 4095;
    int ct = r >> 8;
    int mt = r & 255;
    int hi   = lane >> 4;
    int lcol = lane & 15;
    int m    = mt * 16 + lcol;                       // output column index
    int n_w  = m >> 6;                               // m = n_w*64 + w_m
    int w_m  = m & 63;
    const __bf16* attb  = att  + (size_t)b * NHW * NHW;
    const __bf16* attPb = attP + (size_t)b * NHW * NHW;
    size_t mrow = (size_t)m * NHW;

    v8f accH = {}, accV = {}, accD = {}, accA = {};
    for (int nk = 0; nk < 128; ++nk) {
        size_t vbase = ((size_t)((b * 16 + ct) * 128 + nk) * 32 + lane) * 16;
        v16bf vf  = *(const v16bf*)(vA  + vbase);
        v16bf vtf = *(const v16bf*)(vTA + vbase);
        int kbase = nk * 32 + hi * 16;               // 16-aligned; stays in one 64-block
        int kdiv  = kbase >> 6;                      // h for p_d, w for p_a
        int nh0   = kbase & 63;                      // n_h base (multiple of 16)
        // p_h / p_v: B[K=key, N=m] = att[m][key]
        v16bf bh = *(const v16bf*)(attb + mrow + kbase);
        // p_d: B[(h,nh),(n_w,w)] = att[nh*64+n_w][h*64+w]  -> attP[n_w][h=kdiv][w_m][nh0..+15]
        v16bf bd = *(const v16bf*)(attPb + ((size_t)((n_w * 64 + kdiv) * 64 + w_m) * 64) + nh0);
        // p_a: B[(w,nh),(n_w,h)] = att[nh*64+n_w][h*64+w]  -> attP[n_w][h=w_m][w=kdiv][nh0..+15]
        v16bf ba = *(const v16bf*)(attPb + ((size_t)((n_w * 64 + w_m) * 64 + kdiv) * 64) + nh0);
        if (nk < 127) __builtin_prefetch((const void*)(attb + mrow + kbase + 32), 0, 3);
        accH = wmma_bf16(vf,  bh, accH);
        accV = wmma_bf16(vtf, bh, accV);
        accD = wmma_bf16(vf,  bd, accD);
        accA = wmma_bf16(vf,  ba, accA);
    }
    // store P^T (m-major, cc contiguous) so the final GEMM's B-fragments are vector loads
    size_t obase = ((size_t)b * NHW + m) * 1024;
#pragma unroll
    for (int rr = 0; rr < 8; ++rr) {
        int cl = ct * 16 + rr + 8 * hi;
        PT[obase + 0 * 256 + cl] = (__bf16)accH[rr];
        PT[obase + 1 * 256 + cl] = (__bf16)accV[rr];
        PT[obase + 2 * 256 + cl] = (__bf16)accD[rr];
        PT[obase + 3 * 256 + cl] = (__bf16)accA[rr];
    }
}

// ======= output projection + epilogue: out = gamma*(wo x P + bo) + x =============
__global__ __launch_bounds__(256) void k_proj(const __bf16* __restrict__ woA,
        const __bf16* __restrict__ PT, const float* __restrict__ bo,
        const float* __restrict__ gamma, const float* __restrict__ x,
        float* __restrict__ out) {
    int wid  = threadIdx.x >> 5;
    int lane = threadIdx.x & 31;
    int gw = blockIdx.x * 8 + wid;                   // NB*16*256 = 16384 waves
    int b  = gw >> 12;
    int r  = gw & 4095;
    int ot = r >> 8;
    int mt = r & 255;
    int hi   = lane >> 4;
    int m    = mt * 16 + (lane & 15);
    v8f acc = {};
    size_t prow = ((size_t)b * NHW + m) * 1024 + hi * 16;
#pragma unroll 4
    for (int kk = 0; kk < 32; ++kk) {
        v16bf af = *(const v16bf*)(woA + ((size_t)(ot * 32 + kk) * 32 + lane) * 16);
        v16bf bf = *(const v16bf*)(PT + prow + kk * 32);
        acc = wmma_bf16(af, bf, acc);
    }
    float g = gamma[0];
#pragma unroll
    for (int rr = 0; rr < 8; ++rr) {
        int o = ot * 16 + rr + 8 * hi;
        size_t idx = ((size_t)b * NC + o) * NHW + m;
        out[idx] = g * (acc[rr] + bo[o]) + x[idx];
    }
}

// ================================= launcher ======================================
extern "C" void kernel_launch(void* const* d_in, const int* in_sizes, int n_in,
                              void* d_out, int out_size, void* d_ws, size_t ws_size,
                              hipStream_t stream) {
    const float* x     = (const float*)d_in[0];
    const float* wq    = (const float*)d_in[1];
    const float* bq    = (const float*)d_in[2];
    const float* wk    = (const float*)d_in[3];
    const float* bk    = (const float*)d_in[4];
    const float* wv    = (const float*)d_in[5];
    const float* bv    = (const float*)d_in[6];
    const float* wo    = (const float*)d_in[7];
    const float* bo    = (const float*)d_in[8];
    const float* gamma = (const float*)d_in[9];
    float* out = (float*)d_out;

    // workspace layout (bytes); all region offsets 1MB-aligned
    char* ws = (char*)d_ws;
    __bf16* qA   = (__bf16*)(ws + 0);                 //   1 MB : 4*256*512
    __bf16* kB   = (__bf16*)(ws + (1u   << 20));      //   1 MB
    __bf16* woA  = (__bf16*)(ws + (2u   << 20));      //   0.5 MB : 16*32*512
    __bf16* vA   = (__bf16*)(ws + (3u   << 20));      //   8 MB : 4*16*128*512
    __bf16* vTA  = (__bf16*)(ws + (11u  << 20));      //   8 MB
    __bf16* PT   = (__bf16*)(ws + (19u  << 20));      //  32 MB : 4*4096*1024
    __bf16* att  = (__bf16*)(ws + (51u  << 20));      // 128 MB : 4*4096*4096
    __bf16* attP = (__bf16*)(ws + (179u << 20));      // 128 MB : permuted copy
    // total = 307 MB

    k_prep_q <<<2048,  256, 0, stream>>>(x, wq, bq, qA);
    k_prep_k <<<2048,  256, 0, stream>>>(x, wk, bk, kB);
    k_prep_v <<<16384, 256, 0, stream>>>(x, wv, bv, vA, vTA);
    k_prep_wo<<<1024,  256, 0, stream>>>(wo, woA);
    k_energy_softmax<<<8192, 256, 0, stream>>>(qA, kB, att, attP);
    k_aggregate     <<<2048, 256, 0, stream>>>(vA, vTA, att, attP, PT);
    k_proj          <<<2048, 256, 0, stream>>>(woA, PT, bo, gamma, x, out);
}
